// Mixer2dTriU_30640296690078
// MI455X (gfx1250) — matrix operand order repaired and checked by reading, not compiled, not run on hardware
//
#include <hip/hip_runtime.h>
#include <hip/hip_bf16.h>
#include <stdint.h>

// Problem dims
#define B_DIM 128
#define T_DIM 512
#define C_DIM 512
#define H_DIM 1024   // 2*C

typedef __attribute__((ext_vector_type(16))) __bf16 v16bf;
typedef __attribute__((ext_vector_type(8)))  float  v8f;

// ---------- bf16 helpers: hardware RNE convert ----------
__device__ __forceinline__ unsigned short f2bf_bits(float f) {
    union { __bf16 b; unsigned short s; } o;
    o.b = (__bf16)f;                      // v_cvt to bf16, round-to-nearest-even
    return o.s;
}

// ---------- GEMM tiling ----------
#define BM 128
#define BN 128
#define BK 32
#define LDK 40   // Bst row stride (elems): 80B -> 16B aligned, conflict-free frag reads

// A fragment (16x32 bf16, contiguous K, not ISA table order) loaded straight from a row-major [M][ld]
// bf16 matrix in global memory. Per lane: two 16B contiguous runs.
// lanes 0-15: M=lane, K=0..15 ; lanes 16-31: K=16..31.
__device__ __forceinline__ v16bf frag_a_g(const __bf16* base, long ld, long mrow0, int kb, int lane) {
    const int hl = (lane >> 4) & 1;
    const __bf16* p = base + (mrow0 + (lane & 15)) * ld + kb + hl * 16;
    union { uint4 u[2]; v16bf v; } t;
    t.u[0] = *reinterpret_cast<const uint4*>(p);        // K = kb+16hl   .. +7
    t.u[1] = *reinterpret_cast<const uint4*>(p + 8);    // K = kb+16hl+8 .. +7
    return t.v;
}

// B fragment (32x16 bf16 KxN) from a K-contiguous-per-column matrix: base[n*ld + k].
// lanes 0-15: N=lane, K=0..15 ; lanes 16-31: N=lane-16, K=16..31  -> one 32B run.
__device__ __forceinline__ v16bf frag_b_g(const __bf16* base, long ld, long ncol0, int kb, int lane) {
    const int hl = (lane >> 4) & 1;
    const __bf16* p = base + (ncol0 + (lane & 15)) * ld + kb + hl * 16;
    union { uint4 u[2]; v16bf v; } t;
    t.u[0] = *reinterpret_cast<const uint4*>(p);
    t.u[1] = *reinterpret_cast<const uint4*>(p + 8);
    return t.v;
}

// Same B fragment but from LDS, Bst[n][k] with row stride LDK.
__device__ __forceinline__ v16bf frag_b_lds(const __bf16* Bst, int nbase, int lane) {
    const int hl = (lane >> 4) & 1;
    const __bf16* p = Bst + (nbase + (lane & 15)) * LDK + hl * 16;
    union { uint4 u[2]; v16bf v; } t;
    t.u[0] = *reinterpret_cast<const uint4*>(p);
    t.u[1] = *reinterpret_cast<const uint4*>(p + 8);
    return t.v;
}

// ---------------------------------------------------------------------------
// Kernel 1a: Wm = tril(triu_W) cast to bf16
// ---------------------------------------------------------------------------
__global__ void maskw_kernel(const float* __restrict__ W, unsigned short* __restrict__ Wm) {
    const int idx = blockIdx.x * blockDim.x + threadIdx.x;  // grid covers T*T
    const int s = idx / T_DIM;
    const int t = idx - s * T_DIM;
    const float v = (t <= s) ? W[idx] : 0.0f;
    Wm[idx] = f2bf_bits(v);
}

// ---------------------------------------------------------------------------
// Kernel 1b: plain fp32 -> bf16 cast (for fc1_W / fc2_W), float4 vectorized
// ---------------------------------------------------------------------------
__global__ void castbf_kernel(const float* __restrict__ src, unsigned short* __restrict__ dst) {
    const int i = blockIdx.x * blockDim.x + threadIdx.x;   // grid covers count/4
    const float4 v = reinterpret_cast<const float4*>(src)[i];
    ushort4 p;
    p.x = f2bf_bits(v.x); p.y = f2bf_bits(v.y);
    p.z = f2bf_bits(v.z); p.w = f2bf_bits(v.w);
    reinterpret_cast<ushort4*>(dst)[i] = p;
}

// ---------------------------------------------------------------------------
// Kernel 2/4: joint LayerNorm over (T,C) per batch. One block per batch.
// outF (optional, may alias `in`): fp32. outBF (optional): bf16 bits.
// ---------------------------------------------------------------------------
__global__ __launch_bounds__(1024)
void ln2d_kernel(const float* in, const float* __restrict__ w,
                 const float* __restrict__ bia, float* outF,
                 unsigned short* __restrict__ outBF) {
    const int  N    = T_DIM * C_DIM;
    const int  N4   = N / 4;
    const long base = (long)blockIdx.x * N;
    const int  tid  = threadIdx.x;

    const float4* in4 = reinterpret_cast<const float4*>(in + base);
    float s = 0.f, ss = 0.f;
    for (int i = tid; i < N4; i += 1024) {
        const float4 v = in4[i];
        s  += v.x + v.y + v.z + v.w;
        ss += v.x * v.x + v.y * v.y + v.z * v.z + v.w * v.w;
    }
#pragma unroll
    for (int off = 16; off > 0; off >>= 1) {   // wave32 reduce
        s  += __shfl_xor(s, off, 32);
        ss += __shfl_xor(ss, off, 32);
    }
    __shared__ float rS[32], rSS[32];
    __shared__ float sMu, sRs;
    const int wid = tid >> 5, lane = tid & 31;
    if (lane == 0) { rS[wid] = s; rSS[wid] = ss; }
    __syncthreads();
    if (tid == 0) {
        float S = 0.f, SS = 0.f;
        for (int i = 0; i < 32; ++i) { S += rS[i]; SS += rSS[i]; }
        const float mu  = S / (float)N;
        const float var = SS / (float)N - mu * mu;
        sMu = mu;
        sRs = rsqrtf(var + 1e-5f);
    }
    __syncthreads();
    const float mu = sMu, rs = sRs;

    const float4* w4 = reinterpret_cast<const float4*>(w);
    const float4* b4 = reinterpret_cast<const float4*>(bia);
    for (int i = tid; i < N4; i += 1024) {
        const float4 v = in4[i], wv = w4[i], bv = b4[i];
        float4 o;
        o.x = (v.x - mu) * rs * wv.x + bv.x;
        o.y = (v.y - mu) * rs * wv.y + bv.y;
        o.z = (v.z - mu) * rs * wv.z + bv.z;
        o.w = (v.w - mu) * rs * wv.w + bv.w;
        if (outF) reinterpret_cast<float4*>(outF + base)[i] = o;
        if (outBF) {
            ushort4 p;
            p.x = f2bf_bits(o.x); p.y = f2bf_bits(o.y);
            p.z = f2bf_bits(o.z); p.w = f2bf_bits(o.w);
            reinterpret_cast<ushort4*>(outBF + base)[i] = p;
        }
    }
}

// ---------------------------------------------------------------------------
// Kernel 3: time-mix batched GEMM. R[b,s,c] = sum_t Wm[s,t]*x[b,t,c] + tb[s] + inp
// M=T (s), N=C (c), K=T (t). A: direct-global bf16 frags (L2-resident Wm).
// B: x is [K][N] -> stage transposed into LDS Bst[n][k] via V_PERM_B32 packing.
// Causal: Wm[s,t]=0 for t>s, so the K loop stops at m0+BM (halves work on avg).
// ---------------------------------------------------------------------------
__global__ __launch_bounds__(256)
void timemix_kernel(const unsigned short* __restrict__ Wm_,
                    const unsigned short* __restrict__ X_,
                    const float* __restrict__ tb,
                    const float* __restrict__ inp,
                    float* __restrict__ R) {
    __shared__ __bf16 Bst[BN * LDK];   // 128 x 40 bf16 = 10 KB
    const __bf16* Wm = reinterpret_cast<const __bf16*>(Wm_);

    const int  tid  = threadIdx.x;
    const int  lane = tid & 31, wid = tid >> 5;
    const int  wm   = (wid >> 2) * 64;   // 2 waves in M
    const int  wn   = (wid & 3) * 32;    // 4 waves in N
    const int  m0   = blockIdx.y * BM;
    const int  n0   = blockIdx.x * BN;
    const long bb   = (long)blockIdx.z * T_DIM * C_DIM;
    const __bf16* Xb = reinterpret_cast<const __bf16*>(X_) + bb;

    v8f zero = {};
    v8f acc[4][2];
#pragma unroll
    for (int i = 0; i < 4; ++i)
#pragma unroll
        for (int j = 0; j < 2; ++j) acc[i][j] = zero;

    // staging coords: thread reads 2 K-rows x 8 cols (coalesced), writes packed b32
    const int k0 = (tid >> 4) * 2;     // 0,2,..,30
    const int c0 = (tid & 15) * 8;     // 0,8,..,120

    const int kend = m0 + BM;          // causal bound (<= T_DIM)
    for (int kb = 0; kb < kend; kb += BK) {
        {   // stage x tile transposed: Bst[n][k] = x[kb+k][n0+n], V_PERM pack
            const __bf16* g0 = Xb + (long)(kb + k0) * C_DIM + n0 + c0;
            const uint4 r0 = *reinterpret_cast<const uint4*>(g0);
            const uint4 r1 = *reinterpret_cast<const uint4*>(g0 + C_DIM);
            const unsigned* a0 = reinterpret_cast<const unsigned*>(&r0);
            const unsigned* a1 = reinterpret_cast<const unsigned*>(&r1);
#pragma unroll
            for (int d = 0; d < 4; ++d) {
                // lo = {k0+1 elem 2d (hi16), k0 elem 2d (lo16)} ; hi = same for 2d+1
                const unsigned lo = __builtin_amdgcn_perm(a1[d], a0[d], 0x05040100u);
                const unsigned hi = __builtin_amdgcn_perm(a1[d], a0[d], 0x07060302u);
                *reinterpret_cast<unsigned*>(&Bst[(c0 + 2 * d)     * LDK + k0]) = lo;
                *reinterpret_cast<unsigned*>(&Bst[(c0 + 2 * d + 1) * LDK + k0]) = hi;
            }
        }
        if (kb + BK < kend) {   // global_prefetch_b8 next x tile
            __builtin_prefetch(Xb + (long)(kb + BK + k0) * C_DIM + n0 + c0, 0, 0);
        }
        __syncthreads();

        v16bf bf[2];
#pragma unroll
        for (int j = 0; j < 2; ++j) bf[j] = frag_b_lds(Bst, wn + j * 16, lane);
#pragma unroll
        for (int i = 0; i < 4; ++i) {
            const v16bf af = frag_a_g(Wm, T_DIM, m0 + wm + i * 16, kb, lane);
#pragma unroll
            for (int j = 0; j < 2; ++j)
                acc[i][j] = __builtin_amdgcn_wmma_f32_16x16x32_bf16(
                    false, af, false, bf[j], (short)0, acc[i][j], false, false);
        }
        __syncthreads();
    }

    const int hl = lane >> 4, nl = lane & 15;
#pragma unroll
    for (int i = 0; i < 4; ++i)
#pragma unroll
        for (int j = 0; j < 2; ++j) {
            const int ncol = n0 + wn + j * 16 + nl;
#pragma unroll
            for (int r = 0; r < 8; ++r) {
                const int  srow = m0 + wm + i * 16 + r + 8 * hl;
                const long idx  = bb + (long)srow * C_DIM + ncol;
                R[idx] = acc[i][j][r] + tb[srow] + inp[idx];
            }
        }
}

// ---------------------------------------------------------------------------
// Kernel 5: fc1 GEMM + exact GELU. h[m,n]=gelu(sum_k x2[m,k]*W1[n,k]+b[n])
// M=B*T, N=2C, K=C. All operands bf16, fragments loaded directly from
// global (weights + activation panels are L2/L0 resident). No LDS, no barriers.
// ---------------------------------------------------------------------------
__global__ __launch_bounds__(256)
void fc1_kernel(const unsigned short* __restrict__ X2bf_, const unsigned short* __restrict__ W1_,
                const float* __restrict__ bias, unsigned short* __restrict__ Hout) {
    const __bf16* A  = reinterpret_cast<const __bf16*>(X2bf_);
    const __bf16* Wb = reinterpret_cast<const __bf16*>(W1_);

    const int  tid  = threadIdx.x;
    const int  lane = tid & 31, wid = tid >> 5;
    const int  wm   = (wid >> 2) * 64, wn = (wid & 3) * 32;
    const int  n0   = blockIdx.x * BN;          // over H_DIM
    const long m0   = (long)blockIdx.y * BM;    // over B*T

    v8f zero = {};
    v8f acc[4][2];
#pragma unroll
    for (int i = 0; i < 4; ++i)
#pragma unroll
        for (int j = 0; j < 2; ++j) acc[i][j] = zero;

    for (int kb = 0; kb < C_DIM; kb += BK) {
        v16bf bf[2];
#pragma unroll
        for (int j = 0; j < 2; ++j) bf[j] = frag_b_g(Wb, C_DIM, n0 + wn + j * 16, kb, lane);
#pragma unroll
        for (int i = 0; i < 4; ++i) {
            const v16bf af = frag_a_g(A, C_DIM, m0 + wm + i * 16, kb, lane);
#pragma unroll
            for (int j = 0; j < 2; ++j)
                acc[i][j] = __builtin_amdgcn_wmma_f32_16x16x32_bf16(
                    false, af, false, bf[j], (short)0, acc[i][j], false, false);
        }
    }

    const int hl = lane >> 4, nl = lane & 15;
#pragma unroll
    for (int i = 0; i < 4; ++i)
#pragma unroll
        for (int j = 0; j < 2; ++j) {
            const int ncol = n0 + wn + j * 16 + nl;
            const float bv = bias[ncol];
#pragma unroll
            for (int r = 0; r < 8; ++r) {
                const long mrow = m0 + wm + i * 16 + r + 8 * hl;
                const float x = acc[i][j][r] + bv;
                const float g = 0.5f * x * (1.0f + erff(x * 0.70710678118654752f)); // exact gelu
                Hout[mrow * H_DIM + ncol] = f2bf_bits(g);
            }
        }
}

// ---------------------------------------------------------------------------
// Kernel 6: fc2 GEMM + bias + residual. out[m,n]=sum_k h[m,k]*W2[n,k]+b[n]+x2[m,n]
// M=B*T, N=C, K=2C. Direct-global bf16 fragments, no LDS.
// ---------------------------------------------------------------------------
__global__ __launch_bounds__(256)
void fc2_kernel(const unsigned short* __restrict__ H_, const unsigned short* __restrict__ W2_,
                const float* __restrict__ bias, const float* __restrict__ X2,
                float* __restrict__ out) {
    const __bf16* A  = reinterpret_cast<const __bf16*>(H_);
    const __bf16* Wb = reinterpret_cast<const __bf16*>(W2_);

    const int  tid  = threadIdx.x;
    const int  lane = tid & 31, wid = tid >> 5;
    const int  wm   = (wid >> 2) * 64, wn = (wid & 3) * 32;
    const int  n0   = blockIdx.x * BN;          // over C
    const long m0   = (long)blockIdx.y * BM;    // over B*T

    v8f zero = {};
    v8f acc[4][2];
#pragma unroll
    for (int i = 0; i < 4; ++i)
#pragma unroll
        for (int j = 0; j < 2; ++j) acc[i][j] = zero;

    for (int kb = 0; kb < H_DIM; kb += BK) {
        v16bf bf[2];
#pragma unroll
        for (int j = 0; j < 2; ++j) bf[j] = frag_b_g(Wb, H_DIM, n0 + wn + j * 16, kb, lane);
#pragma unroll
        for (int i = 0; i < 4; ++i) {
            const v16bf af = frag_a_g(A, H_DIM, m0 + wm + i * 16, kb, lane);
#pragma unroll
            for (int j = 0; j < 2; ++j)
                acc[i][j] = __builtin_amdgcn_wmma_f32_16x16x32_bf16(
                    false, af, false, bf[j], (short)0, acc[i][j], false, false);
        }
    }

    const int hl = lane >> 4, nl = lane & 15;
#pragma unroll
    for (int i = 0; i < 4; ++i)
#pragma unroll
        for (int j = 0; j < 2; ++j) {
            const int ncol = n0 + wn + j * 16 + nl;
            const float bv = bias[ncol];
#pragma unroll
            for (int r = 0; r < 8; ++r) {
                const long mrow = m0 + wm + i * 16 + r + 8 * hl;
                const long idx  = mrow * C_DIM + ncol;
                out[idx] = acc[i][j][r] + bv + X2[idx];
            }
        }
}

// ---------------------------------------------------------------------------
// Launch
// ---------------------------------------------------------------------------
extern "C" void kernel_launch(void* const* d_in, const int* in_sizes, int n_in,
                              void* d_out, int out_size, void* d_ws, size_t ws_size,
                              hipStream_t stream) {
    const float* inputs = (const float*)d_in[0];
    const float* ln1_w  = (const float*)d_in[1];
    const float* ln1_b  = (const float*)d_in[2];
    const float* ln2_w  = (const float*)d_in[3];
    const float* ln2_b  = (const float*)d_in[4];
    const float* triu_W = (const float*)d_in[5];
    const float* triu_b = (const float*)d_in[6];
    const float* fc1_W  = (const float*)d_in[7];
    const float* fc1_b  = (const float*)d_in[8];
    const float* fc2_W  = (const float*)d_in[9];
    const float* fc2_b  = (const float*)d_in[10];
    float* out = (float*)d_out;

    // workspace layout (all sizes multiples of 256B)
    char* ws = (char*)d_ws;
    const size_t szX  = (size_t)B_DIM * T_DIM * C_DIM * 2;  // x / x2 bf16 (reused)
    const size_t szWm = (size_t)T_DIM * T_DIM * 2;          // tril(W) bf16
    const size_t szW1 = (size_t)H_DIM * C_DIM * 2;          // fc1_W bf16
    const size_t szW2 = (size_t)C_DIM * H_DIM * 2;          // fc2_W bf16
    const size_t szR  = (size_t)B_DIM * T_DIM * C_DIM * 4;  // R / x2 fp32 (in-place LN2)
    unsigned short* xbf = (unsigned short*)(ws);
    unsigned short* Wm  = (unsigned short*)(ws + szX);
    unsigned short* W1b = (unsigned short*)(ws + szX + szWm);
    unsigned short* W2b = (unsigned short*)(ws + szX + szWm + szW1);
    float*          R   = (float*)(ws + szX + szWm + szW1 + szW2);
    unsigned short* h   = (unsigned short*)(ws + szX + szWm + szW1 + szW2 + szR);

    // 1. weight preparation (bf16)
    maskw_kernel<<<(T_DIM * T_DIM) / 256, 256, 0, stream>>>(triu_W, Wm);
    castbf_kernel<<<(H_DIM * C_DIM) / 4 / 256, 256, 0, stream>>>(fc1_W, W1b);
    castbf_kernel<<<(C_DIM * H_DIM) / 4 / 256, 256, 0, stream>>>(fc2_W, W2b);
    // 2. LN1 -> x (bf16)
    ln2d_kernel<<<B_DIM, 1024, 0, stream>>>(inputs, ln1_w, ln1_b, nullptr, xbf);
    // 3. time-mix GEMM + bias + residual -> R (fp32)
    timemix_kernel<<<dim3(C_DIM / BN, T_DIM / BM, B_DIM), 256, 0, stream>>>(Wm, xbf, triu_b, inputs, R);
    // 4. LN2: R <- x2 (fp32, in-place) and x2 bf16 -> xbf (x dead after timemix)
    ln2d_kernel<<<B_DIM, 1024, 0, stream>>>(R, ln2_w, ln2_b, R, xbf);
    // 5. fc1 + GELU -> h (bf16)
    fc1_kernel<<<dim3(H_DIM / BN, (B_DIM * T_DIM) / BM), 256, 0, stream>>>(xbf, W1b, fc1_b, h);
    // 6. fc2 + bias + residual(x2) -> out
    fc2_kernel<<<dim3(C_DIM / BN, (B_DIM * T_DIM) / BM), 256, 0, stream>>>(h, W2b, fc2_b, R, out);
}
